// Model_11192684773891
// MI455X (gfx1250) — matrix operand order repaired and checked by reading, not compiled, not run on hardware
//
#include <hip/hip_runtime.h>
#include <stdint.h>

#define H 128
#define N_USER 100000
#define N_ITEM 50000
#define E_EDGES 640000
#define L_EDGES 200000

typedef __bf16 v16bf __attribute__((ext_vector_type(16)));
typedef float  v8f   __attribute__((ext_vector_type(8)));
typedef unsigned int v8u __attribute__((ext_vector_type(8)));

__device__ __forceinline__ unsigned int f32_to_bf16(float f) {
    unsigned int u = __float_as_uint(f);
    return (u + 0x7FFFu + ((u >> 16) & 1u)) >> 16;      // RNE
}
__device__ __forceinline__ unsigned int pack_bf16x2(float lo, float hi) {
    return f32_to_bf16(lo) | (f32_to_bf16(hi) << 16);
}

// ---------------------------------------------------------------------------
// W^T tile geometry: LDS pitch 68 dwords (272 B) per 128-K bf16 row:
//  - rows 16B-aligned for async b128 stores
//  - 68*l mod 64 = 4*l -> conflict-free B-fragment ds_loads
// The GLOBAL pre-transposed weight copies use the SAME 272 B pitch, so the
// whole 34 KB stage is a linear copy: the async-load INST_OFFSET is added to
// both the global address and the LDS address (ISA 08_async_tensor), letting
// one address pair + one LDS offset + 8 x offset:j*4352 cover the burst.
// ---------------------------------------------------------------------------
#define A_PITCH_U 65    // dword pitch, 128-col bf16 A row (conflict-free)
#define WP_U      68    // dword pitch of W^T rows in LDS
#define WG_PITCH  136   // ushort pitch of W^T rows in global (= WP_U dwords)

__device__ __forceinline__ void stage_WT_async(unsigned sWT_base, int tid,
                                               const unsigned short* WTg) {
    int n = tid >> 4, seg = tid & 15;   // block covers rows 0..15; +16 rows/step
    unsigned lds0 = sWT_base + (unsigned)(n * (WP_U * 4) + seg * 16);
    unsigned long long g0 = (unsigned long long)(WTg + n * WG_PITCH + seg * 8);
    asm volatile(
        "global_load_async_to_lds_b128 %0, %1, off\n\t"
        "global_load_async_to_lds_b128 %0, %1, off offset:4352\n\t"
        "global_load_async_to_lds_b128 %0, %1, off offset:8704\n\t"
        "global_load_async_to_lds_b128 %0, %1, off offset:13056\n\t"
        "global_load_async_to_lds_b128 %0, %1, off offset:17408\n\t"
        "global_load_async_to_lds_b128 %0, %1, off offset:21760\n\t"
        "global_load_async_to_lds_b128 %0, %1, off offset:26112\n\t"
        "global_load_async_to_lds_b128 %0, %1, off offset:30464\n\t"
        :: "v"(lds0), "v"(g0) : "memory");
}
__device__ __forceinline__ void async_wait0() {
    asm volatile("s_wait_asynccnt 0x0" ::: "memory");
}

// ---------------------------------------------------------------------------
// One-shot weight prep: WT[n*WG_PITCH + k] = bf16(W[k*H + n]) for k in [0,128)
// (transpose + convert + pad to the LDS-matching pitch).
// ---------------------------------------------------------------------------
__global__ void weight_prep_kernel(const float* __restrict__ W,
                                   unsigned short* __restrict__ WT) {
    int i = blockIdx.x * 256 + threadIdx.x;   // i over 128*128 elements
    if (i >= H * H) return;
    int n = i >> 7, k = i & 127;
    WT[n * WG_PITCH + k] = (unsigned short)f32_to_bf16(W[k * H + n]);
}

// ---------------------------------------------------------------------------
// Edge scatter: agg[dst] += x[src]; cnt[dst] += 1.   32 lanes per edge, float4.
// agg buffers (77 MB total) are L2-resident (192 MB) -> atomics resolve in L2.
// ---------------------------------------------------------------------------
__global__ void scatter_sum_kernel(const float* __restrict__ x,
                                   const int* __restrict__ src,
                                   const int* __restrict__ dst,
                                   float* __restrict__ agg,
                                   float* __restrict__ cnt,
                                   int nE) {
    int t = blockIdx.x * blockDim.x + threadIdx.x;
    int e = t >> 5;
    if (e >= nE) return;
    int sl = t & 31;
    int s = src[e], d = dst[e];
    float4 v = ((const float4*)x)[(size_t)s * 32 + sl];
    float* ap = agg + (size_t)d * H + sl * 4;
    atomicAdd(ap + 0, v.x);
    atomicAdd(ap + 1, v.y);
    atomicAdd(ap + 2, v.z);
    atomicAdd(ap + 3, v.w);
    if (sl == 0) atomicAdd(cnt + d, 1.0f);
}

// ---------------------------------------------------------------------------
// Fused SAGE layer GEMM:  out = act( (A1/cnt) @ W1  +  A2 @ W2  + bias )
// 256 thr = 8 waves; block computes 16 rows x 128 cols; wave w owns the 16x16
// tile at columns [16w,16w+16) via v_wmma_f32_16x16x32_bf16. Whole W^T per
// pass in one 8-instruction async burst; 4 K-chunk WMMAs back-to-back.
// ---------------------------------------------------------------------------
template <bool RELU>
__global__ void sage_gemm_kernel(const float* __restrict__ A1,
                                 const float* __restrict__ cnt,
                                 const unsigned short* __restrict__ WT1g,
                                 const float* __restrict__ A2,
                                 const unsigned short* __restrict__ WT2g,
                                 const float* __restrict__ bias,
                                 float* __restrict__ out) {
    const int tid  = threadIdx.x;
    const int lane = tid & 31;
    const int wave = tid >> 5;           // 0..7 => column tile
    const int row0 = blockIdx.x * 16;

    __shared__ __align__(16) unsigned int sA1[16 * A_PITCH_U];
    __shared__ __align__(16) unsigned int sA2[16 * A_PITCH_U];
    __shared__ __align__(16) unsigned int sWT[128 * WP_U];
    const unsigned sWT_base = (unsigned)(unsigned long long)(void*)sWT;

    // Stage A tiles (16 rows x 64 bf16 pairs each); scale A1 by 1/max(cnt,1).
    for (int p = tid; p < 1024; p += 256) {
        int r = p >> 6, kp = p & 63;
        int row = row0 + r;
        float sc = 1.0f / fmaxf(cnt[row], 1.0f);
        float2 a1 = ((const float2*)A1)[(size_t)row * 64 + kp];
        float2 a2 = ((const float2*)A2)[(size_t)row * 64 + kp];
        sA1[r * A_PITCH_U + kp] = pack_bf16x2(a1.x * sc, a1.y * sc);
        sA2[r * A_PITCH_U + kp] = pack_bf16x2(a2.x, a2.y);
    }

    const int r  = lane & 15;
    const int hi = lane >> 4;
    const float bval = bias[wave * 16 + r];
    v8f acc = {};

    for (int pass = 0; pass < 2; ++pass) {
        const unsigned int* sAu = pass ? sA2 : sA1;
        if (pass) __syncthreads();           // pass-0 sWT reads finished
        stage_WT_async(sWT_base, tid, pass ? WT2g : WT1g);
        async_wait0();
        __syncthreads();                     // W staged (and A on pass 0)
#pragma unroll
        for (int kc = 0; kc < 4; ++kc) {
            // A frag: lanes 0-15 K 0-15, lanes 16-31 K 16-31
            v8u au, bu;
            int ab = r * A_PITCH_U + kc * 16 + hi * 8;
            au[0] = sAu[ab + 0]; au[1] = sAu[ab + 1];
            au[2] = sAu[ab + 2]; au[3] = sAu[ab + 3];
            au[4] = sAu[ab + 4]; au[5] = sAu[ab + 5];
            au[6] = sAu[ab + 6]; au[7] = sAu[ab + 7];
            // B frag: lanes 0-15 K 0-15, lanes 16-31 K 16-31, N = lane&15
            int bb = (wave * 16 + r) * WP_U + kc * 16 + hi * 8;
#pragma unroll
            for (int v = 0; v < 8; ++v) bu[v] = sWT[bb + v];
            acc = __builtin_amdgcn_wmma_f32_16x16x32_bf16(
                false, __builtin_bit_cast(v16bf, au),
                false, __builtin_bit_cast(v16bf, bu),
                (short)0, acc, false, false);
        }
    }

    // C/D layout: VGPR v -> M = v + 8*hi, N = wave*16 + (lane&15)
    const int ncol = wave * 16 + r;
#pragma unroll
    for (int v = 0; v < 8; ++v) {
        float o = acc[v] + bval;
        if (RELU) o = fmaxf(o, 0.0f);
        out[(size_t)(row0 + hi * 8 + v) * H + ncol] = o;
    }
}

// ---------------------------------------------------------------------------
// Fused edge decoder: per block of 16 edges:
//   A = [zU[el_row] ; zI[el_col]]  (16 x 256, gathered, bf16 in LDS)
//   z1 = relu(A @ W1 + b1)  (16 x 128, WMMA, 8 waves)  -> LDS
//   out = z1 @ W2 + b2      (length-128 dot per edge)
// W_lin1^T staged as two pitch-matched 34 KB async halves.
// ---------------------------------------------------------------------------
#define AD_PITCH_U 130  // dword pitch for 256-col bf16 A row

__global__ void edge_decoder_kernel(const float* __restrict__ zU,
                                    const float* __restrict__ zI,
                                    const int* __restrict__ el_row,
                                    const int* __restrict__ el_col,
                                    const unsigned short* __restrict__ WT1g,
                                    const float* __restrict__ b1,
                                    const float* __restrict__ W2,
                                    const float* __restrict__ b2,
                                    float* __restrict__ out) {
    const int tid  = threadIdx.x;
    const int lane = tid & 31;
    const int wave = tid >> 5;
    const int e0   = blockIdx.x * 16;

    __shared__ __align__(16) unsigned int sA[16 * AD_PITCH_U];
    __shared__ __align__(16) unsigned int sWT[128 * WP_U];
    __shared__ float sZ1[16 * 128];
    __shared__ float sW2[128];
    __shared__ float red[256];
    __shared__ int sRow[16], sCol[16];
    const unsigned sWT_base = (unsigned)(unsigned long long)(void*)sWT;

    if (tid < 16) { sRow[tid] = el_row[e0 + tid]; sCol[tid] = el_col[e0 + tid]; }
    if (tid < 128) sW2[tid] = W2[tid];
    __syncthreads();

    // Gather + concat A: 16 rows x 128 bf16 pairs (first 64 pairs user, rest item)
    for (int p = tid; p < 2048; p += 256) {
        int r = p >> 7, kp = p & 127;
        float2 v = (kp < 64)
                       ? ((const float2*)zU)[(size_t)sRow[r] * 64 + kp]
                       : ((const float2*)zI)[(size_t)sCol[r] * 64 + (kp - 64)];
        sA[r * AD_PITCH_U + kp] = pack_bf16x2(v.x, v.y);
    }

    const int r  = lane & 15;
    const int hi = lane >> 4;
    const float bval = b1[wave * 16 + r];
    v8f acc = {};

    for (int h = 0; h < 2; ++h) {            // two K-halves of W_lin1^T
        if (h) __syncthreads();
        stage_WT_async(sWT_base, tid, WT1g + h * (128 * WG_PITCH));
        async_wait0();
        __syncthreads();
#pragma unroll
        for (int c = 0; c < 4; ++c) {
            int kc = h * 4 + c;
            v8u au, bu;
            int ab = r * AD_PITCH_U + kc * 16 + hi * 8;
            au[0] = sA[ab + 0]; au[1] = sA[ab + 1];
            au[2] = sA[ab + 2]; au[3] = sA[ab + 3];
            au[4] = sA[ab + 4]; au[5] = sA[ab + 5];
            au[6] = sA[ab + 6]; au[7] = sA[ab + 7];
            int bb = (wave * 16 + r) * WP_U + c * 16 + hi * 8;
#pragma unroll
            for (int v = 0; v < 8; ++v) bu[v] = sWT[bb + v];
            acc = __builtin_amdgcn_wmma_f32_16x16x32_bf16(
                false, __builtin_bit_cast(v16bf, au),
                false, __builtin_bit_cast(v16bf, bu),
                (short)0, acc, false, false);
        }
    }

    // z1 -> LDS with bias + relu
    {
        int ncol = wave * 16 + r;
#pragma unroll
        for (int v = 0; v < 8; ++v)
            sZ1[(hi * 8 + v) * 128 + ncol] = fmaxf(acc[v] + bval, 0.0f);
    }
    __syncthreads();

    // out[e] = dot(z1[e,:], W2) + b2 :  16 edges x 16 segments of 8
    int e = tid >> 4, seg = tid & 15;
    float partial = 0.0f;
#pragma unroll
    for (int j = 0; j < 8; ++j)
        partial += sZ1[e * 128 + seg * 8 + j] * sW2[seg * 8 + j];
    red[tid] = partial;
    __syncthreads();
    if (seg < 8) red[tid] += red[tid + 8];
    __syncthreads();
    if (seg < 4) red[tid] += red[tid + 4];
    __syncthreads();
    if (seg < 2) red[tid] += red[tid + 2];
    __syncthreads();
    if (seg == 0) out[e0 + e] = red[tid] + red[tid + 1] + b2[0];
}

// ---------------------------------------------------------------------------
extern "C" void kernel_launch(void* const* d_in, const int* in_sizes, int n_in,
                              void* d_out, int out_size, void* d_ws, size_t ws_size,
                              hipStream_t stream) {
    (void)in_sizes; (void)n_in; (void)out_size; (void)ws_size;
    const float* x_user = (const float*)d_in[0];
    const float* x_item = (const float*)d_in[1];
    const float* Wl1_ui = (const float*)d_in[2];
    const float* Wr1_ui = (const float*)d_in[3];
    const float* b1_ui  = (const float*)d_in[4];
    const float* Wl1_iu = (const float*)d_in[5];
    const float* Wr1_iu = (const float*)d_in[6];
    const float* b1_iu  = (const float*)d_in[7];
    const float* Wl2_ui = (const float*)d_in[8];
    const float* Wr2_ui = (const float*)d_in[9];
    const float* b2_ui  = (const float*)d_in[10];
    const float* Wl2_iu = (const float*)d_in[11];
    const float* Wr2_iu = (const float*)d_in[12];
    const float* b2_iu  = (const float*)d_in[13];
    const float* W_lin1 = (const float*)d_in[14];
    const float* b_lin1 = (const float*)d_in[15];
    const float* W_lin2 = (const float*)d_in[16];
    const float* b_lin2 = (const float*)d_in[17];
    const int* e_src = (const int*)d_in[18];
    const int* e_dst = (const int*)d_in[19];
    const int* el_row = (const int*)d_in[20];
    const int* el_col = (const int*)d_in[21];
    float* out = (float*)d_out;

    // Workspace layout (z buffers alias agg buffers: safe, A staged in LDS
    // before the same rows are overwritten, blocks own disjoint rows).
    float* ws   = (float*)d_ws;
    float* aggI = ws;                                  // N_ITEM*H  (also zI)
    float* aggU = aggI + (size_t)N_ITEM * H;           // N_USER*H  (also zU)
    float* hI   = aggU + (size_t)N_USER * H;           // N_ITEM*H
    float* hU   = hI   + (size_t)N_ITEM * H;           // N_USER*H
    float* cntI = hU   + (size_t)N_USER * H;           // N_ITEM
    float* cntU = cntI + N_ITEM;                       // N_USER
    // bf16 transposed weights, pitch-136 rows (16B-aligned: prefix 154,200,000 B;
    // each matrix 128*136 ushorts = 34,816 B)
    const size_t WTSZ = 128 * WG_PITCH;                // ushorts per matrix
    unsigned short* wt = (unsigned short*)(cntU + N_USER);
    unsigned short* wtWl1_ui = wt + 0 * WTSZ;
    unsigned short* wtWr1_ui = wt + 1 * WTSZ;
    unsigned short* wtWl1_iu = wt + 2 * WTSZ;
    unsigned short* wtWr1_iu = wt + 3 * WTSZ;
    unsigned short* wtWl2_ui = wt + 4 * WTSZ;
    unsigned short* wtWr2_ui = wt + 5 * WTSZ;
    unsigned short* wtWl2_iu = wt + 6 * WTSZ;
    unsigned short* wtWr2_iu = wt + 7 * WTSZ;
    unsigned short* wtLin1   = wt + 8 * WTSZ;          // two halves: 2*WTSZ

    const dim3 blk(256);
    const int scatterGrid = (E_EDGES * 32 + 255) / 256;

    // ---- one-shot weight transpose+convert to padded bf16 ----
    weight_prep_kernel<<<64, blk, 0, stream>>>(Wl1_ui, wtWl1_ui);
    weight_prep_kernel<<<64, blk, 0, stream>>>(Wr1_ui, wtWr1_ui);
    weight_prep_kernel<<<64, blk, 0, stream>>>(Wl1_iu, wtWl1_iu);
    weight_prep_kernel<<<64, blk, 0, stream>>>(Wr1_iu, wtWr1_iu);
    weight_prep_kernel<<<64, blk, 0, stream>>>(Wl2_ui, wtWl2_ui);
    weight_prep_kernel<<<64, blk, 0, stream>>>(Wr2_ui, wtWr2_ui);
    weight_prep_kernel<<<64, blk, 0, stream>>>(Wl2_iu, wtWl2_iu);
    weight_prep_kernel<<<64, blk, 0, stream>>>(Wr2_iu, wtWr2_iu);
    weight_prep_kernel<<<64, blk, 0, stream>>>(W_lin1, wtLin1);                 // K-half 0
    weight_prep_kernel<<<64, blk, 0, stream>>>(W_lin1 + 128 * H, wtLin1 + WTSZ); // K-half 1

    // ---- layer 1 aggregation (aggI,aggU contiguous; cntI,cntU contiguous) ----
    hipMemsetAsync(aggI, 0, (size_t)(N_ITEM + N_USER) * H * sizeof(float), stream);
    hipMemsetAsync(cntI, 0, (size_t)(N_ITEM + N_USER) * sizeof(float), stream);
    scatter_sum_kernel<<<scatterGrid, blk, 0, stream>>>(x_user, e_src, e_dst, aggI, cntI, E_EDGES);
    scatter_sum_kernel<<<scatterGrid, blk, 0, stream>>>(x_item, e_dst, e_src, aggU, cntU, E_EDGES);

    // ---- layer 1 fused GEMMs (+relu) ----
    sage_gemm_kernel<true><<<N_ITEM / 16, blk, 0, stream>>>(aggI, cntI, wtWl1_ui, x_item, wtWr1_ui, b1_ui, hI);
    sage_gemm_kernel<true><<<N_USER / 16, blk, 0, stream>>>(aggU, cntU, wtWl1_iu, x_user, wtWr1_iu, b1_iu, hU);

    // ---- layer 2 aggregation ----
    hipMemsetAsync(aggI, 0, (size_t)(N_ITEM + N_USER) * H * sizeof(float), stream);
    hipMemsetAsync(cntI, 0, (size_t)(N_ITEM + N_USER) * sizeof(float), stream);
    scatter_sum_kernel<<<scatterGrid, blk, 0, stream>>>(hU, e_src, e_dst, aggI, cntI, E_EDGES);
    scatter_sum_kernel<<<scatterGrid, blk, 0, stream>>>(hI, e_dst, e_src, aggU, cntU, E_EDGES);

    // ---- layer 2 fused GEMMs (no relu); outputs alias agg buffers ----
    sage_gemm_kernel<false><<<N_ITEM / 16, blk, 0, stream>>>(aggI, cntI, wtWl2_ui, hI, wtWr2_ui, b2_ui, aggI);
    sage_gemm_kernel<false><<<N_USER / 16, blk, 0, stream>>>(aggU, cntU, wtWl2_iu, hU, wtWr2_iu, b2_iu, aggU);

    // ---- fused edge decoder ----
    edge_decoder_kernel<<<L_EDGES / 16, blk, 0, stream>>>(aggU, aggI, el_row, el_col,
                                                          wtLin1, b_lin1, W_lin2, b_lin2, out);
}
